// MutualInformation_67293547594316
// MI455X (gfx1250) — matrix operand order repaired and checked by reading, not compiled, not run on hardware
//
#include <hip/hip_runtime.h>
#include <math.h>

// ---------------------------------------------------------------------------
// Mutual information via KDE histograms on MI455X (gfx1250).
//   joint = kv1^T @ kv2  : batched GEMM, M=N=256, K=65536, B=8  (~68.7 GFLOP)
//   computed with v_wmma_f32_16x16x32_f16; kv tiles generated *sparsely*
//   (sigma=0.1 -> each pixel hits <=4 bins; exp(-50 r^2) underflows beyond).
// ---------------------------------------------------------------------------

typedef __attribute__((ext_vector_type(16))) _Float16 v16h;
typedef __attribute__((ext_vector_type(8)))  _Float16 v8h;
typedef __attribute__((ext_vector_type(8)))  float    v8f;

#define NB     256
#define BATCH  8
#define HW     65536
#define TM     128      // joint tile rows  (bins of image1)
#define TN     128      // joint tile cols  (bins of image2)
#define TK     128      // pixels staged per LDS step
#define KSPLIT 8        // K-dimension split across workgroups
#define EPS    1e-10f
#define COEF  (-50.0f)  // -0.5 / sigma^2, sigma = 0.1

union AFrag { v16h v; v8h h[2]; };

// ---------------------------------------------------------------------------
__global__ void __launch_bounds__(256)
mi_zero_kernel(float* __restrict__ ws, int n) {
  int i = blockIdx.x * blockDim.x + threadIdx.x;
  if (i < n) ws[i] = 0.0f;
}

// ---------------------------------------------------------------------------
// Marginal kernel sums: marg[b][bin] = sum_pixels exp(-50 (v-bin)^2).
// Sparse: <=4 bins per pixel, accumulated with ds_add_f32 in LDS.
__global__ void __launch_bounds__(256)
mi_marginal_kernel(const float* __restrict__ x1, const float* __restrict__ x2,
                   float* __restrict__ marg1, float* __restrict__ marg2) {
  __shared__ float h[2 * NB];
  const int t  = threadIdx.x;
  const int sp = blockIdx.x;          // pixel split 0..7
  const int b  = blockIdx.y;          // batch

  for (int i = t; i < 2 * NB; i += 256) h[i] = 0.0f;
  __syncthreads();

  const int chunk = HW / 8;
  for (int i = t; i < chunk; i += 256) {
    int p = b * HW + sp * chunk + i;
    float v1 = 255.0f * x1[p];
    float v2 = 255.0f * x2[p];
    int i1 = (int)floorf(v1), i2 = (int)floorf(v2);
#pragma unroll
    for (int d = -1; d <= 2; ++d) {
      int b1 = i1 + d; float r1 = v1 - (float)b1;
      if (b1 >= 0 && b1 < NB) atomicAdd(&h[b1],      __expf(COEF * r1 * r1));
      int b2 = i2 + d; float r2 = v2 - (float)b2;
      if (b2 >= 0 && b2 < NB) atomicAdd(&h[NB + b2], __expf(COEF * r2 * r2));
    }
  }
  __syncthreads();
  atomicAdd(&marg1[b * NB + t], h[t]);
  atomicAdd(&marg2[b * NB + t], h[NB + t]);
}

// ---------------------------------------------------------------------------
// Joint histogram via WMMA. Grid: (KSPLIT, 4 tiles [mT,nT], BATCH).
// Each WG: 128x128 tile of joint[b], K-chunk of 8192 pixels, accumulated
// into global with atomic f32 adds (partial-K tiles sum linearly).
__global__ void __launch_bounds__(256)
mi_joint_wmma_kernel(const float* __restrict__ x1, const float* __restrict__ x2,
                     float* __restrict__ joint) {
  __shared__ __align__(16) _Float16 tiles[2 * TM * TK];  // 64 KB of 320 KB LDS
  _Float16* At = tiles;            // [m][k] row-major: bin-of-img1 x pixel
  _Float16* Bt = tiles + TM * TK;  // [n][k]           : bin-of-img2 x pixel

  const int t  = threadIdx.x;
  const int ks = blockIdx.x;
  const int mT = blockIdx.y >> 1;
  const int nT = blockIdx.y & 1;
  const int b  = blockIdx.z;

  const int w    = t >> 5, lane = t & 31;  // 8 waves of 32 (wave32)
  const int wm   = w & 1,  wn   = w >> 1;  // 2x4 wave grid: 64x32 per wave
  const int lrow = lane & 15, lhi = lane >> 4;

  v8f acc[4][2];
  {
    v8f z = {};
#pragma unroll
    for (int mi = 0; mi < 4; ++mi)
#pragma unroll
      for (int ni = 0; ni < 2; ++ni) acc[mi][ni] = z;
  }

  // fill role: threads 0..127 own pixel columns of A, 128..255 of B
  const int img  = t >> 7;
  const int col  = t & (TK - 1);
  const float* __restrict__ src = img ? x2 : x1;
  _Float16* dst  = img ? Bt : At;
  const int bin0 = (img ? nT : mT) * TM;

  const int kChunk = HW / KSPLIT;           // 8192
  const int kBase  = b * HW + ks * kChunk;
  const int nSteps = kChunk / TK;           // 64

  for (int step = 0; step < nSteps; ++step) {
    const int k0 = kBase + step * TK;

    __syncthreads();  // previous step's WMMA reads complete
    // zero both tiles with ds_store_b128 (16 stores/thread)
    {
      uint4  z4 = {0u, 0u, 0u, 0u};
      uint4* zp = (uint4*)tiles;
#pragma unroll
      for (int i = 0; i < (2 * TM * TK) / (8 * 256); ++i)
        zp[t + i * 256] = z4;
    }
    __syncthreads();

    // sparse fill: <=4 nonzero bins per pixel column
    {
      float v = 255.0f * src[k0 + col];
      if (step + 1 < nSteps) __builtin_prefetch(src + k0 + col + TK, 0, 3);
      int ib = (int)floorf(v);
#pragma unroll
      for (int d = -1; d <= 2; ++d) {
        int   bn = ib + d;
        float r  = v - (float)bn;
        int   lm = bn - bin0;
        if (bn >= 0 && bn < NB && lm >= 0 && lm < TM)
          dst[lm * TK + col] = (_Float16)__expf(COEF * r * r);
      }
    }
    __syncthreads();

    // 128x128 += 128x128-K MAC: 8 v_wmma_f32_16x16x32_f16 per wave per k-step
#pragma unroll
    for (int kk = 0; kk < TK / 32; ++kk) {
      AFrag bf[2];
#pragma unroll
      for (int ni = 0; ni < 2; ++ni) {
        // B 32x16 layout: lane -> col lrow, 16 contiguous K at lhi*16
        const _Float16* Bp = &Bt[(wn * 32 + ni * 16 + lrow) * TK + kk * 32 + lhi * 16];
        bf[ni].h[0] = *(const v8h*)Bp;
        bf[ni].h[1] = *(const v8h*)(Bp + 8);
      }
#pragma unroll
      for (int mi = 0; mi < 4; ++mi) {
        // A 16x32 layout: lane -> row lrow, 16 contiguous K at lhi*16
        const _Float16* Ap = &At[(wm * 64 + mi * 16 + lrow) * TK + kk * 32 + lhi * 16];
        AFrag af;
        af.h[0] = *(const v8h*)Ap;
        af.h[1] = *(const v8h*)(Ap + 8);
#pragma unroll
        for (int ni = 0; ni < 2; ++ni)
          acc[mi][ni] = __builtin_amdgcn_wmma_f32_16x16x32_f16(
              false, af.v, false, bf[ni].v, (short)0, acc[mi][ni], false, false);
      }
    }
  }

  // merge partial 128x128 tile into joint[b] (C layout: M = r + lhi*8, N = lrow)
#pragma unroll
  for (int mi = 0; mi < 4; ++mi)
#pragma unroll
    for (int ni = 0; ni < 2; ++ni)
#pragma unroll
      for (int r = 0; r < 8; ++r) {
        int Mg = mT * TM + wm * 64 + mi * 16 + lhi * 8 + r;
        int Ng = nT * TN + wn * 32 + ni * 16 + lrow;
        atomicAdd(&joint[(b * NB + Mg) * NB + Ng], acc[mi][ni][r]);
      }
}

// ---------------------------------------------------------------------------
__device__ __forceinline__ float block_reduce_sum(float v, float* red, int t) {
  red[t] = v;
  __syncthreads();
  for (int s = 128; s > 0; s >>= 1) {
    if (t < s) red[t] += red[t + s];
    __syncthreads();
  }
  float r = red[0];
  __syncthreads();
  return r;  // broadcast
}

__global__ void __launch_bounds__(256)
mi_finalize_kernel(const float* __restrict__ joint,
                   const float* __restrict__ marg1,
                   const float* __restrict__ marg2,
                   float* __restrict__ out) {
  __shared__ float red[256];
  const int t = threadIdx.x, b = blockIdx.x;

  // marginal entropies; pdf = mean over HW then /(sum+EPS)  =>  p = m/(S + EPS*HW)
  float m1 = marg1[b * NB + t];
  float S1 = block_reduce_sum(m1, red, t);
  float p1 = m1 / (S1 + EPS * (float)HW);
  float H1 = block_reduce_sum(-p1 * __log2f(p1 + EPS), red, t);

  float m2 = marg2[b * NB + t];
  float S2 = block_reduce_sum(m2, red, t);
  float p2 = m2 / (S2 + EPS * (float)HW);
  float H2 = block_reduce_sum(-p2 * __log2f(p2 + EPS), red, t);

  // joint entropy
  float js = 0.0f;
  for (int i = t; i < NB * NB; i += 256) js += joint[b * NB * NB + i];
  float JN = block_reduce_sum(js, red, t) + EPS;

  float hj = 0.0f;
  for (int i = t; i < NB * NB; i += 256) {
    float pj = joint[b * NB * NB + i] / JN;
    hj -= pj * __log2f(pj + EPS);
  }
  float H12 = block_reduce_sum(hj, red, t);

  if (t == 0) {
    float mi = H1 + H2 - H12;
    out[b] = 2.0f * mi / (H1 + H2);   // NORMALIZE=True
  }
}

// ---------------------------------------------------------------------------
extern "C" void kernel_launch(void* const* d_in, const int* in_sizes, int n_in,
                              void* d_out, int out_size, void* d_ws, size_t ws_size,
                              hipStream_t stream) {
  (void)in_sizes; (void)n_in; (void)out_size; (void)ws_size;

  const float* x1 = (const float*)d_in[0];   // (8,1,256,256) f32
  const float* x2 = (const float*)d_in[1];   // (8,1,256,256) f32
  // d_in[2] = bins: linspace(0,255,256) == the integers 0..255, folded into math.

  float* out   = (float*)d_out;              // (8,) f32
  float* joint = (float*)d_ws;               // 8*256*256 f32
  float* marg1 = joint + BATCH * NB * NB;    // 8*256 f32
  float* marg2 = marg1 + BATCH * NB;         // 8*256 f32

  const int nzero = BATCH * NB * NB + 2 * BATCH * NB;
  mi_zero_kernel<<<(nzero + 255) / 256, 256, 0, stream>>>(joint, nzero);

  mi_marginal_kernel<<<dim3(8, BATCH), 256, 0, stream>>>(x1, x2, marg1, marg2);

  mi_joint_wmma_kernel<<<dim3(KSPLIT, 4, BATCH), 256, 0, stream>>>(x1, x2, joint);

  mi_finalize_kernel<<<BATCH, 256, 0, stream>>>(joint, marg1, marg2, out);
}
